// HGTSpatialEncoder_71012989272521
// MI455X (gfx1250) — matrix operand order repaired and checked by reading, not compiled, not run on hardware
//
#include <hip/hip_runtime.h>
#include <math.h>

#define N_NODES 10000
#define N_EDGES 320000
#define DIN 128
#define HID 256
#define DOUTD 128
#define NH 4

typedef __bf16 bf16_t;
typedef __attribute__((ext_vector_type(16))) bf16_t v16bf;
typedef __attribute__((ext_vector_type(8)))  bf16_t bf16x8;
typedef __attribute__((ext_vector_type(4)))  bf16_t bf16x4;
typedef __attribute__((ext_vector_type(8)))  float  v8f;
typedef __attribute__((ext_vector_type(4)))  unsigned int u32x4;
typedef __attribute__((ext_vector_type(8)))  int i32x8;
typedef __attribute__((ext_vector_type(4)))  int i32x4;

#if defined(__has_builtin)
#if __has_builtin(__builtin_amdgcn_tensor_load_to_lds)
#define HAVE_TDM 1
#endif
#endif
#ifndef HAVE_TDM
#define HAVE_TDM 0
#endif

// round-to-nearest-even float -> bf16
__device__ inline bf16_t f2bf(float f) {
    unsigned u = __float_as_uint(f);
    u += 0x7FFFu + ((u >> 16) & 1u);
    unsigned short h = (unsigned short)(u >> 16);
    return __builtin_bit_cast(bf16_t, h);
}
__device__ inline float bf2f(bf16_t b) {
    unsigned short s = __builtin_bit_cast(unsigned short, b);
    return __uint_as_float((unsigned)s << 16);
}

// ---------------------------------------------------------------------------
// WMMA GEMM: C[M,O] = epilogue(A[M,K] @ W[O,K]^T + bias), A/W bf16, acc f32.
// MODE 0: bias   MODE 1: bias+relu
// MODE 3: relu(beta*(val+bias) + (1-beta)*skip), beta = sigmoid(*gate)
// OUT bit0: write Cf (f32);  OUT bit1: write Cb (bf16)
// One wave = one 16x16 tile; block = 4 waves = 16 rows x 64 cols.
// A tile (16 x K bf16) staged in LDS once per block via TDM
// (tensor_load_to_lds with 16B row padding to dodge bank conflicts).
// ---------------------------------------------------------------------------
template<int MODE, int OUT>
__global__ __launch_bounds__(128)
void gemm_bf16(const bf16_t* __restrict__ A, const bf16_t* __restrict__ W,
               const float* __restrict__ bias, float* __restrict__ Cf,
               bf16_t* __restrict__ Cb, int M, int K, int O,
               const float* __restrict__ skipbuf, const float* __restrict__ gate)
{
    __shared__ __align__(16) unsigned char sA[16 * (2 * 256 + 16)];  // max K=256

    const int wave = threadIdx.x >> 5;
    const int lane = threadIdx.x & 31;
    const int m    = lane & 15;
    const int sel  = lane >> 4;
    const int row0 = blockIdx.x * 16;
    const int col0 = blockIdx.y * 64 + wave * 16;
    const int rowbytes = 2 * K + 16;     // padded LDS row stride

#if HAVE_TDM
    if (wave == 0) {
        const unsigned long long ga = (unsigned long long)(uintptr_t)(A + (size_t)row0 * K);
        const unsigned lds_off = (unsigned)(uintptr_t)sA;   // flat->LDS offset = low 32 bits
        const int padInt = __builtin_ctz((unsigned)(K >> 1)) - 1;  // K/2 DWORDs per row
        u32x4 g0;
        g0[0] = 1u;                                              // count=1, user desc
        g0[1] = lds_off;                                         // lds_addr
        g0[2] = (unsigned)(ga & 0xffffffffull);                  // global_addr[31:0]
        g0[3] = (unsigned)((ga >> 32) & 0x01ffffffull) | 0x80000000u;  // addr[56:32] | type=2
        i32x8 g1;
        g1[0] = (1 << 16)             // data_size = 2B
              | (1 << 20)             // pad_enable
              | (padInt << 22)        // pad_interval = one row of DWORDs
              | (3 << 25);            // pad_amount = 4 DWORDs (16B)
        g1[1] = (K & 0xffff) << 16;                 // tensor_dim0[15:0]
        g1[2] = ((unsigned)K >> 16) | (16u << 16);  // tensor_dim0[31:16] | tensor_dim1=16
        g1[3] = (K << 16);                          // tile_dim0 = K
        g1[4] = 16;                                 // tile_dim1 = 16
        g1[5] = K;                                  // tensor_dim0_stride
        g1[6] = 0;
        g1[7] = 0;
        i32x4 gz = {0, 0, 0, 0};
#if defined(__clang_major__) && __clang_major__ >= 23
        i32x8 gz8 = {0, 0, 0, 0, 0, 0, 0, 0};
        __builtin_amdgcn_tensor_load_to_lds(g0, g1, gz, gz, gz8, 0);
#else
        __builtin_amdgcn_tensor_load_to_lds(g0, g1, gz, gz, 0);
#endif
#if __has_builtin(__builtin_amdgcn_s_wait_tensorcnt)
        __builtin_amdgcn_s_wait_tensorcnt(0);
#else
        asm volatile("s_wait_tensorcnt 0x0" ::: "memory");
#endif
    }
    __syncthreads();
#else
    {
        const int rowChunks = K >> 3;                    // 16B chunks per row
        const int chunks = rowChunks * 16;
        for (int c = threadIdx.x; c < chunks; c += 128) {
            int r  = c / rowChunks;
            int cc = c - r * rowChunks;
            uint4 vv = ((const uint4*)(A + (size_t)(row0 + r) * K))[cc];
            *(uint4*)(sA + r * rowbytes + cc * 16) = vv;
        }
    }
    __syncthreads();
#endif

    const bf16_t* __restrict__ wrow = W + (size_t)(col0 + m) * K;
    const unsigned char* arow = sA + m * rowbytes;

    v8f acc = {0.f, 0.f, 0.f, 0.f, 0.f, 0.f, 0.f, 0.f};

    for (int kb = 0; kb < K; kb += 32) {
        __builtin_prefetch(wrow + kb + 256, 0, 0);
        // A fragment from LDS: 16 contiguous k (k = 16*sel + s), matching B
        const bf16x8 alo = *(const bf16x8*)(arow + 2 * (kb + sel * 16));
        const bf16x8 ahi = *(const bf16x8*)(arow + 2 * (kb + sel * 16 + 8));
        const v16bf af = __builtin_shufflevector(alo, ahi,
            0, 1, 2, 3, 4, 5, 6, 7, 8, 9, 10, 11, 12, 13, 14, 15);
        // B fragment from global: 16 contiguous bf16 of W row (= B column)
        const v16bf bfrag = *(const v16bf*)(wrow + kb + sel * 16);

        acc = __builtin_amdgcn_wmma_f32_16x16x32_bf16(
                  false, af, false, bfrag, (short)0, acc, false, false);
    }

    float beta = 1.f;
    if (MODE == 3) beta = 1.f / (1.f + expf(-gate[0]));
    const int col = col0 + m;
    const float bv = bias[col];
#pragma unroll
    for (int j = 0; j < 8; ++j) {
        const int row = row0 + j + sel * 8;
        float v = acc[j] + bv;
        if (MODE == 1) v = fmaxf(v, 0.f);
        if (MODE == 3) {
            float s = skipbuf[(size_t)row * O + col];
            v = fmaxf(beta * v + (1.f - beta) * s, 0.f);
        }
        if (OUT & 1) Cf[(size_t)row * O + col] = v;
        if (OUT & 2) Cb[(size_t)row * O + col] = f2bf(v);
    }
}

// ---------------------------------------------------------------------------
// Fold relation transform into projection weight (bf16 weight out, f32 bias):
//   Wdst[h*D+e, c] = sum_d rel[h,d,e] * Wsrc[h*D+d, c]
// ---------------------------------------------------------------------------
__global__ void fold_rel(const float* __restrict__ Wsrc, const float* __restrict__ bsrc,
                         const float* __restrict__ rel, bf16_t* __restrict__ Wdst,
                         float* __restrict__ bdst, int OUTD, int K, int D)
{
    int tid = blockIdx.x * blockDim.x + threadIdx.x;
    int total = OUTD * (K + 1);
    if (tid >= total) return;
    int o = tid / (K + 1);
    int c = tid - o * (K + 1);
    int h = o / D, e = o - h * D;
    const float* relh = rel + (size_t)h * D * D;
    float s = 0.f;
    if (c < K) {
        for (int d = 0; d < D; ++d) s += relh[d * D + e] * Wsrc[(size_t)(h * D + d) * K + c];
        Wdst[(size_t)o * K + c] = f2bf(s);
    } else {
        for (int d = 0; d < D; ++d) s += relh[d * D + e] * bsrc[h * D + d];
        bdst[o] = s;
    }
}

__global__ void cvt_bf16(const float* __restrict__ x, bf16_t* __restrict__ y, int n)
{
    int i = blockIdx.x * blockDim.x + threadIdx.x;
    if (i < n) y[i] = f2bf(x[i]);
}

__global__ void fill_f(float* __restrict__ p, float v, int n)
{
    int i = blockIdx.x * blockDim.x + threadIdx.x;
    if (i < n) p[i] = v;
}

__global__ void gelu_bf(const float* __restrict__ x, bf16_t* __restrict__ y, int n)
{
    int i = blockIdx.x * blockDim.x + threadIdx.x;
    if (i < n) {
        float v = x[i];
        y[i] = f2bf(0.5f * v * (1.f + erff(v * 0.70710678118654752f)));
    }
}

// alpha[e,h] = (q[dst,h,:]·k[src,h,:]) * p[h]/sqrt(D);  atomic segment-max over dst
// q/k stored bf16 (gather bandwidth halved); dot accumulated in f32.
__global__ void edge_logits(const bf16_t* __restrict__ q, const bf16_t* __restrict__ k,
                            const int* __restrict__ src, const int* __restrict__ dst,
                            const float* __restrict__ p, float inv_sqrt_d,
                            float* __restrict__ alpha, float* __restrict__ amax,
                            int E, int D, int OD)
{
    int tid = blockIdx.x * blockDim.x + threadIdx.x;
    if (tid >= E * NH) return;
    int e = tid / NH, h = tid - e * NH;
    int s = src[e], d = dst[e];
    const bf16_t* qp = q + (size_t)d * OD + h * D;
    const bf16_t* kp = k + (size_t)s * OD + h * D;
    float acc = 0.f;
    for (int i = 0; i < D; i += 8) {
        bf16x8 qq = *(const bf16x8*)(qp + i);
        bf16x8 kk = *(const bf16x8*)(kp + i);
#pragma unroll
        for (int j = 0; j < 8; ++j) acc += bf2f(qq[j]) * bf2f(kk[j]);
    }
    float a = acc * p[h] * inv_sqrt_d;
    alpha[tid] = a;
    float* ap = amax + (size_t)d * NH + h;     // pre-filled with -inf
    if (a >= 0.f) atomicMax((int*)ap, __float_as_int(a));
    else          atomicMin((unsigned int*)ap, __float_as_uint(a));
}

__global__ void edge_exp(float* __restrict__ alpha, const float* __restrict__ amax,
                         const int* __restrict__ dst, float* __restrict__ den, int E)
{
    int tid = blockIdx.x * blockDim.x + threadIdx.x;
    if (tid >= E * NH) return;
    int e = tid / NH, h = tid - e * NH;
    int d = dst[e];
    float ex = expf(alpha[tid] - amax[(size_t)d * NH + h]);
    alpha[tid] = ex;
    atomicAdd(&den[(size_t)d * NH + h], ex);
}

// agg[dst,h,:] += w * v[src,h,:]; v gathered in bf16, accumulation f32 atomics
__global__ void edge_scatter(const bf16_t* __restrict__ vbuf, const float* __restrict__ ex,
                             const float* __restrict__ den, const int* __restrict__ src,
                             const int* __restrict__ dst, float* __restrict__ agg,
                             int E, int D, int OD)
{
    int Dq = D >> 2;
    int per_edge = NH * Dq;
    int tid = blockIdx.x * blockDim.x + threadIdx.x;
    if (tid >= E * per_edge) return;
    int e  = tid / per_edge;
    int r  = tid - e * per_edge;
    int h  = r / Dq;
    int d4 = (r - h * Dq) << 2;
    int s = src[e], d = dst[e];
    float w = ex[(size_t)e * NH + h] / (den[(size_t)d * NH + h] + 1e-16f);
    const bf16x4 vv = *(const bf16x4*)(vbuf + (size_t)s * OD + h * D + d4);
    float* out = agg + (size_t)d * OD + h * D + d4;
    atomicAdd(out + 0, bf2f(vv[0]) * w);
    atomicAdd(out + 1, bf2f(vv[1]) * w);
    atomicAdd(out + 2, bf2f(vv[2]) * w);
    atomicAdd(out + 3, bf2f(vv[3]) * w);
}

// ---------------------------------------------------------------------------
extern "C" void kernel_launch(void* const* d_in, const int* in_sizes, int n_in,
                              void* d_out, int out_size, void* d_ws, size_t ws_size,
                              hipStream_t stream)
{
    (void)in_sizes; (void)n_in; (void)out_size; (void)ws_size;
    const float* x     = (const float*)d_in[0];
    const int*   ei    = (const int*)  d_in[1];
    const float* W_in  = (const float*)d_in[2];
    const float* b_in  = (const float*)d_in[3];
    const float* Wk1   = (const float*)d_in[4];
    const float* bk1   = (const float*)d_in[5];
    const float* Wq1   = (const float*)d_in[6];
    const float* bq1   = (const float*)d_in[7];
    const float* Wv1   = (const float*)d_in[8];
    const float* bv1   = (const float*)d_in[9];
    const float* a1    = (const float*)d_in[10];
    const float* m1    = (const float*)d_in[11];
    const float* p1    = (const float*)d_in[12];
    const float* Wa1   = (const float*)d_in[13];
    const float* ba1   = (const float*)d_in[14];
    const float* skip1 = (const float*)d_in[15];
    const float* Wk2   = (const float*)d_in[16];
    const float* bk2   = (const float*)d_in[17];
    const float* Wq2   = (const float*)d_in[18];
    const float* bq2   = (const float*)d_in[19];
    const float* Wv2   = (const float*)d_in[20];
    const float* bv2   = (const float*)d_in[21];
    const float* a2    = (const float*)d_in[22];
    const float* m2    = (const float*)d_in[23];
    const float* p2    = (const float*)d_in[24];
    const float* Wa2   = (const float*)d_in[25];
    const float* ba2   = (const float*)d_in[26];

    char* base = (char*)d_ws;
    size_t off = 0;
    auto alloc = [&](size_t bytes) { void* p = base + off; off += (bytes + 255) & ~size_t(255); return p; };

    float*  h0f  = (float*)alloc((size_t)N_NODES * HID * 4);
    float*  agg  = (float*)alloc((size_t)N_NODES * HID * 4);
    float*  alf  = (float*)alloc((size_t)N_EDGES * NH * 4);
    float*  amax = (float*)alloc((size_t)N_NODES * NH * 4);
    float*  den  = (float*)alloc((size_t)N_NODES * NH * 4);
    bf16_t* qb   = (bf16_t*)alloc((size_t)N_NODES * HID * 2);
    bf16_t* kbuf = (bf16_t*)alloc((size_t)N_NODES * HID * 2);
    bf16_t* vb   = (bf16_t*)alloc((size_t)N_NODES * HID * 2);
    bf16_t* xbf  = (bf16_t*)alloc((size_t)N_NODES * DIN * 2);
    bf16_t* h0b  = (bf16_t*)alloc((size_t)N_NODES * HID * 2);
    bf16_t* h1b  = (bf16_t*)alloc((size_t)N_NODES * HID * 2);
    bf16_t* gbb  = (bf16_t*)alloc((size_t)N_NODES * HID * 2);
    bf16_t* Winb = (bf16_t*)alloc((size_t)HID * DIN * 2);
    bf16_t* Wq1b = (bf16_t*)alloc((size_t)HID * HID * 2);
    bf16_t* Wa1b = (bf16_t*)alloc((size_t)HID * HID * 2);
    bf16_t* Wq2b = (bf16_t*)alloc((size_t)DOUTD * HID * 2);
    bf16_t* Wa2b = (bf16_t*)alloc((size_t)DOUTD * DOUTD * 2);
    bf16_t* Wk1f = (bf16_t*)alloc((size_t)HID * HID * 2);   float* bk1f = (float*)alloc(HID * 4);
    bf16_t* Wv1f = (bf16_t*)alloc((size_t)HID * HID * 2);   float* bv1f = (float*)alloc(HID * 4);
    bf16_t* Wk2f = (bf16_t*)alloc((size_t)DOUTD * HID * 2); float* bk2f = (float*)alloc(DOUTD * 4);
    bf16_t* Wv2f = (bf16_t*)alloc((size_t)DOUTD * HID * 2); float* bv2f = (float*)alloc(DOUTD * 4);

    const int* src = ei;
    const int* dst = ei + N_EDGES;
    const int rowTiles = N_NODES / 16;   // 625

    // bf16 conversions of GEMM operands
    cvt_bf16<<<(N_NODES * DIN + 255) / 256, 256, 0, stream>>>(x, xbf, N_NODES * DIN);
    cvt_bf16<<<(HID * DIN + 255) / 256, 256, 0, stream>>>(W_in, Winb, HID * DIN);
    cvt_bf16<<<(HID * HID + 255) / 256, 256, 0, stream>>>(Wq1, Wq1b, HID * HID);
    cvt_bf16<<<(HID * HID + 255) / 256, 256, 0, stream>>>(Wa1, Wa1b, HID * HID);
    cvt_bf16<<<(DOUTD * HID + 255) / 256, 256, 0, stream>>>(Wq2, Wq2b, DOUTD * HID);
    cvt_bf16<<<(DOUTD * DOUTD + 255) / 256, 256, 0, stream>>>(Wa2, Wa2b, DOUTD * DOUTD);

    // Fold relation transforms into K/V projection weights
    {
        int t1 = HID * (HID + 1), t2 = DOUTD * (HID + 1);
        fold_rel<<<(t1 + 255) / 256, 256, 0, stream>>>(Wk1, bk1, a1, Wk1f, bk1f, HID, HID, 64);
        fold_rel<<<(t1 + 255) / 256, 256, 0, stream>>>(Wv1, bv1, m1, Wv1f, bv1f, HID, HID, 64);
        fold_rel<<<(t2 + 255) / 256, 256, 0, stream>>>(Wk2, bk2, a2, Wk2f, bk2f, DOUTD, HID, 32);
        fold_rel<<<(t2 + 255) / 256, 256, 0, stream>>>(Wv2, bv2, m2, Wv2f, bv2f, DOUTD, HID, 32);
    }

    // h0 = relu(x @ W_in^T + b_in): f32 (skip source) + bf16 (GEMM input)
    gemm_bf16<1, 3><<<dim3(rowTiles, HID / 64), 128, 0, stream>>>(
        xbf, Winb, b_in, h0f, h0b, N_NODES, DIN, HID, nullptr, nullptr);

    // ---- layer 1 (D=64) ----
    gemm_bf16<0, 2><<<dim3(rowTiles, HID / 64), 128, 0, stream>>>(
        h0b, Wq1b, bq1, nullptr, qb, N_NODES, HID, HID, nullptr, nullptr);
    gemm_bf16<0, 2><<<dim3(rowTiles, HID / 64), 128, 0, stream>>>(
        h0b, Wk1f, bk1f, nullptr, kbuf, N_NODES, HID, HID, nullptr, nullptr);
    gemm_bf16<0, 2><<<dim3(rowTiles, HID / 64), 128, 0, stream>>>(
        h0b, Wv1f, bv1f, nullptr, vb, N_NODES, HID, HID, nullptr, nullptr);

    fill_f<<<(N_NODES * NH + 255) / 256, 256, 0, stream>>>(amax, -INFINITY, N_NODES * NH);
    fill_f<<<(N_NODES * NH + 255) / 256, 256, 0, stream>>>(den, 0.f, N_NODES * NH);
    fill_f<<<(N_NODES * HID + 255) / 256, 256, 0, stream>>>(agg, 0.f, N_NODES * HID);

    edge_logits<<<(N_EDGES * NH + 255) / 256, 256, 0, stream>>>(
        qb, kbuf, src, dst, p1, 0.125f, alf, amax, N_EDGES, 64, HID);
    edge_exp<<<(N_EDGES * NH + 255) / 256, 256, 0, stream>>>(alf, amax, dst, den, N_EDGES);
    edge_scatter<<<(N_EDGES * NH * 16 + 255) / 256, 256, 0, stream>>>(
        vb, alf, den, src, dst, agg, N_EDGES, 64, HID);

    gelu_bf<<<(N_NODES * HID + 255) / 256, 256, 0, stream>>>(agg, gbb, N_NODES * HID);
    // h1 = relu(beta*(gelu @ Wa1^T + ba1) + (1-beta)*h0)  -> bf16 only (feeds GEMMs)
    gemm_bf16<3, 2><<<dim3(rowTiles, HID / 64), 128, 0, stream>>>(
        gbb, Wa1b, ba1, nullptr, h1b, N_NODES, HID, HID, h0f, skip1);

    // ---- layer 2 (D=32, out=128) ----
    gemm_bf16<0, 2><<<dim3(rowTiles, DOUTD / 64), 128, 0, stream>>>(
        h1b, Wq2b, bq2, nullptr, qb, N_NODES, HID, DOUTD, nullptr, nullptr);
    gemm_bf16<0, 2><<<dim3(rowTiles, DOUTD / 64), 128, 0, stream>>>(
        h1b, Wk2f, bk2f, nullptr, kbuf, N_NODES, HID, DOUTD, nullptr, nullptr);
    gemm_bf16<0, 2><<<dim3(rowTiles, DOUTD / 64), 128, 0, stream>>>(
        h1b, Wv2f, bv2f, nullptr, vb, N_NODES, HID, DOUTD, nullptr, nullptr);

    fill_f<<<(N_NODES * NH + 255) / 256, 256, 0, stream>>>(amax, -INFINITY, N_NODES * NH);
    fill_f<<<(N_NODES * NH + 255) / 256, 256, 0, stream>>>(den, 0.f, N_NODES * NH);
    fill_f<<<(N_NODES * DOUTD + 255) / 256, 256, 0, stream>>>(agg, 0.f, N_NODES * DOUTD);

    edge_logits<<<(N_EDGES * NH + 255) / 256, 256, 0, stream>>>(
        qb, kbuf, src, dst, p2, 0.17677669529663689f, alf, amax, N_EDGES, 32, DOUTD);
    edge_exp<<<(N_EDGES * NH + 255) / 256, 256, 0, stream>>>(alf, amax, dst, den, N_EDGES);
    edge_scatter<<<(N_EDGES * NH * 8 + 255) / 256, 256, 0, stream>>>(
        vb, alf, den, src, dst, agg, N_EDGES, 32, DOUTD);

    gelu_bf<<<(N_NODES * DOUTD + 255) / 256, 256, 0, stream>>>(agg, gbb, N_NODES * DOUTD);
    gemm_bf16<0, 1><<<dim3(rowTiles, DOUTD / 64), 128, 0, stream>>>(
        gbb, Wa2b, ba2, (float*)d_out, nullptr, N_NODES, DOUTD, DOUTD, nullptr, nullptr);
}
